// PaiNerf_2723009266475
// MI455X (gfx1250) — matrix operand order repaired and checked by reading, not compiled, not run on hardware
//
#include <hip/hip_runtime.h>
#include <hip/hip_bf16.h>

typedef _Float16 h16;
typedef __attribute__((ext_vector_type(16))) _Float16 v16h;
typedef __attribute__((ext_vector_type(8)))  _Float16 v8h;
typedef __attribute__((ext_vector_type(8)))  float    v8f;

#define PAD32(x) (((x) + 31) & ~31)

union AFrag { v16h v; v8h h[2]; };

// A fragment (16x32 f16): lane<16 -> M=lane, K = {kb..kb+7} U {kb+16..kb+23};
// lane>=16 -> M=lane-16, K = {kb+8..kb+15} U {kb+24..kb+31}.  Both runs contiguous.
__device__ __forceinline__ v16h load_frag_a(const h16* buf, int stride, int rb, int kb, int lane) {
  const int m = lane & 15, hf = lane >> 4;
  const h16* p = buf + (size_t)(rb + m) * stride + kb + hf * 8;
  AFrag f;
  f.h[0] = *(const v8h*)(p);
  f.h[1] = *(const v8h*)(p + 16);
  return f.v;
}

// B fragment (32x16 f16) from pre-transposed weights Wt[n][kp]:
// lane<16 -> K = {kb..kb+7} U {kb+16..kb+23}, lane>=16 -> K = {kb+8..kb+15} U {kb+24..kb+31}; column n = lane&15.
__device__ __forceinline__ v16h load_frag_b(const h16* wt, int kp, int nb, int kb, int lane) {
  const int n = lane & 15, hf = lane >> 4;
  const h16* p = wt + (size_t)(nb + n) * kp + kb + hf * 8;
  AFrag f;
  f.h[0] = *(const v8h*)(p);
  f.h[1] = *(const v8h*)(p + 16);
  return f.v;
}

struct MlpW {
  const h16*   Wt[8];    // pre-transposed f16 weights, row stride = K[l]
  const float* bias[8];  // fp32 biases (device)
  int          K[8];     // padded K per layer
  const h16*   WoutT;    // (16, W) f16, zero-padded columns
  const float* bout;
};

// One hidden layer with CG-wide column blocking:
// dst[m][doff+n] = relu(src[m][:K] @ Wt + bias).
// Each wave keeps one A fragment in registers for CG WMMAs (CG independent
// accumulator chains -> latency hiding + 1/CG the LDS traffic).
template<int TILE, int CG>
__device__ __forceinline__ void layer_mm(const h16* src, int sstride, int K,
                                         h16* dst, int dstride, int doff, int NW,
                                         const h16* wt, const float* bias, int tid) {
  const int lane = tid & 31, wave = tid >> 5;
  const int RT = TILE / 16;
  const int ntasks = RT * (NW / (16 * CG));
  for (int t = wave; t < ntasks; t += 4) {
    const int rb = (t % RT) * 16, nbg = (t / RT) * 16 * CG;
    v8f acc[CG];
#pragma unroll
    for (int c = 0; c < CG; ++c) acc[c] = {};
    for (int kb = 0; kb < K; kb += 32) {
      v16h a = load_frag_a(src, sstride, rb, kb, lane);
#pragma unroll
      for (int c = 0; c < CG; ++c) {
        v16h b = load_frag_b(wt, K, nbg + c * 16, kb, lane);
        acc[c] = __builtin_amdgcn_wmma_f32_16x16x32_f16(false, a, false, b, (short)0, acc[c],
                                                        false, false);
      }
    }
    const int n = lane & 15, hf = lane >> 4;
#pragma unroll
    for (int c = 0; c < CG; ++c) {
      const float bv = bias[nbg + c * 16 + n];
#pragma unroll
      for (int r = 0; r < 8; ++r) {
        float v = acc[c][r] + bv;           // C layout: VGPR r -> M = r + 8*hf
        v = v > 0.f ? v : 0.f;
        dst[(size_t)(rb + r + hf * 8) * dstride + doff + nbg + c * 16 + n] = (h16)v;
      }
    }
  }
}

// Run D hidden layers with skip-concat after layer SKIP (h written into BUF at col IN).
template<int TILE, int CG, int IN, int W, int D, int SKIP, int CONCP>
__device__ __forceinline__ const h16* run_hidden(h16* BUF, h16* H0, h16* H1,
                                                 const MlpW& mw, int tid) {
  const h16* src = BUF; int sstride = CONCP;
  h16* ha = H0; h16* hb = H1;
  for (int l = 0; l < D; ++l) {
    const bool sk = (l == SKIP);
    h16* dst = sk ? BUF : ha;
    layer_mm<TILE, CG>(src, sstride, mw.K[l], dst, sk ? CONCP : W, sk ? IN : 0, W,
                       mw.Wt[l], mw.bias[l], tid);
    __syncthreads();
    if (sk) { src = BUF; sstride = CONCP; }
    else    { src = ha; sstride = W; h16* t = ha; ha = hb; hb = t; }
  }
  return src;  // final layer is never the skip layer -> stride W
}

// ---------------- prepass: posenc(coords) and verts @ f1W + f1b ----------------
__global__ void prep_kernel(const float* __restrict__ coords, const float* __restrict__ verts,
                            const float* __restrict__ f1W, const float* __restrict__ f1b,
                            float* __restrict__ c40, float* __restrict__ vf, int N) {
  int v = blockIdx.x * blockDim.x + threadIdx.x;
  if (v >= N) return;
  float x0 = coords[(size_t)v * 2 + 0], x1 = coords[(size_t)v * 2 + 1];
  float fr = 1.f;
  for (int f = 0; f < 10; ++f) {
    c40[(size_t)v * 40 + f * 4 + 0] = __sinf(x0 * fr);
    c40[(size_t)v * 40 + f * 4 + 1] = __sinf(x1 * fr);
    c40[(size_t)v * 40 + f * 4 + 2] = __cosf(x0 * fr);
    c40[(size_t)v * 40 + f * 4 + 3] = __cosf(x1 * fr);
    fr *= 2.f;
  }
  for (int b = 0; b < 2; ++b) {
    const float* p = verts + ((size_t)b * N + v) * 3;
    float v0 = p[0], v1 = p[1], v2 = p[2];
    for (int o = 0; o < 40; ++o)
      vf[((size_t)b * N + v) * 40 + o] = v0 * f1W[o] + v1 * f1W[40 + o] + v2 * f1W[80 + o] + f1b[o];
  }
}

// ------------- prepass: fp32 (K,N) weights -> f16 transposed (Ndst, Kp) -------------
struct ConvJob { const float* src; h16* dst; int K, Kp, Nsrc, Ndst; };
struct ConvJobs { ConvJob j[36]; };

__global__ void convert_weights(ConvJobs jobs) {
  ConvJob jb = jobs.j[blockIdx.x];
  const int tot = jb.Ndst * jb.Kp;
  for (int i = threadIdx.x; i < tot; i += blockDim.x) {
    int n = i / jb.Kp, k = i % jb.Kp;
    float v = (k < jb.K && n < jb.Nsrc) ? jb.src[(size_t)k * jb.Nsrc + n] : 0.f;
    jb.dst[(size_t)n * jb.Kp + k] = (h16)v;
  }
}

// ---------------- encoder: x = [c40(40), vf(40)] -> lat(8) ----------------
template<int W, int D, int SKIP>
__global__ __launch_bounds__(128) void enc_kernel(const float* __restrict__ c40,
                                                  const float* __restrict__ vf,
                                                  MlpW mw, float* __restrict__ lat,
                                                  int seg_start, int seg_len, int N) {
  constexpr int IN = 80, TILE = 32, OUT = 8;
  constexpr int CG = (W >= 256) ? 4 : 2;
  constexpr int CONCP = PAD32(IN + W);
  __shared__ __align__(16) h16 BUF[TILE * CONCP];
  __shared__ __align__(16) h16 HB[2][TILE * W];
  const int tid = threadIdx.x;
  const int base = blockIdx.x * TILE;
  const int total = 2 * seg_len;

  for (int i = tid; i < TILE * CONCP; i += 128) BUF[i] = (h16)0.f;
  __syncthreads();
  for (int i = tid; i < TILE * IN; i += 128) {
    int m = i / IN, f = i % IN;
    int pi = base + m;
    if (pi < total) {
      int b = (pi >= seg_len) ? 1 : 0;            // B == 2: avoid runtime division
      int v = seg_start + pi - b * seg_len;
      float val = (f < 40) ? c40[(size_t)v * 40 + f]
                           : vf[((size_t)b * N + v) * 40 + (f - 40)];
      BUF[m * CONCP + f] = (h16)val;
    }
  }
  __syncthreads();

  const h16* src = run_hidden<TILE, CG, IN, W, D, SKIP, CONCP>(BUF, HB[0], HB[1], mw, tid);

  const int lane = tid & 31, wave = tid >> 5;
  for (int t = wave; t < TILE / 16; t += 4) {
    const int rb = t * 16;
    v8f acc = {};
    for (int kb = 0; kb < W; kb += 32) {
      v16h a = load_frag_a(src, W, rb, kb, lane);
      v16h b = load_frag_b(mw.WoutT, W, 0, kb, lane);
      acc = __builtin_amdgcn_wmma_f32_16x16x32_f16(false, a, false, b, (short)0, acc, false, false);
    }
    const int n = lane & 15, hf = lane >> 4;
    if (n < OUT) {
      const float bv = mw.bout[n];
#pragma unroll
      for (int r = 0; r < 8; ++r) {
        int pi = base + rb + r + hf * 8;
        if (pi < total) {
          int b = (pi >= seg_len) ? 1 : 0;
          int v = seg_start + pi - b * seg_len;
          lat[((size_t)b * N + v) * 8 + n] = acc[r] + bv;
        }
      }
    }
  }
}

// -------- decoder: din = [fv(8), bc(3), m(40), tc-m(120)] -> out(3), scattered --------
template<int W, int D, int SKIP>
__global__ __launch_bounds__(128) void dec_kernel(const float* __restrict__ lat,
                                                  const float* __restrict__ c40,
                                                  const float* __restrict__ bcoords,
                                                  const int* __restrict__ trilist,
                                                  const float* __restrict__ mmap,
                                                  const int* __restrict__ tmpl,
                                                  MlpW mw, float* __restrict__ out,
                                                  int seg_start, int seg_len, int N, int P) {
  constexpr int IN = 171, TILE = 32, OUT = 3;
  constexpr int CG = (W >= 256) ? 4 : 2;
  constexpr int CONCP = PAD32(IN + W);
  __shared__ __align__(16) h16 BUF[TILE * CONCP];
  __shared__ __align__(16) h16 HB[2][TILE * W];
  const int tid = threadIdx.x;
  const int base = blockIdx.x * TILE;
  const int total = 2 * seg_len;

  for (int i = tid; i < TILE * CONCP; i += 128) BUF[i] = (h16)0.f;
  __syncthreads();
  for (int i = tid; i < TILE * IN; i += 128) {
    int m = i / IN, f = i % IN;
    int pi = base + m;
    if (pi >= total) continue;
    int b = (pi >= seg_len) ? 1 : 0;              // B == 2: avoid runtime division
    int j = pi - b * seg_len;
    int p = seg_start + j;
    float val;
    if (f < 8) {               // barycentric-interpolated latent
      val = 0.f;
      for (int t3 = 0; t3 < 3; ++t3) {
        int vtx = trilist[(size_t)p * 3 + t3];
        val += bcoords[(size_t)p * 3 + t3] * lat[((size_t)b * N + vtx) * 8 + f];
      }
    } else if (f < 11) {       // bcoords
      val = bcoords[(size_t)p * 3 + (f - 8)];
    } else if (f < 51) {       // map
      val = mmap[(size_t)j * 40 + (f - 11)];
    } else {                   // tci = c40[tri[t]] - m, t-major
      int g = f - 51, t3 = g / 40, q = g % 40;
      int vtx = trilist[(size_t)p * 3 + t3];
      val = c40[(size_t)vtx * 40 + q] - mmap[(size_t)j * 40 + q];
    }
    BUF[m * CONCP + f] = (h16)val;
  }
  __syncthreads();

  const h16* src = run_hidden<TILE, CG, IN, W, D, SKIP, CONCP>(BUF, HB[0], HB[1], mw, tid);

  const int lane = tid & 31, wave = tid >> 5;
  for (int t = wave; t < TILE / 16; t += 4) {
    const int rb = t * 16;
    v8f acc = {};
    for (int kb = 0; kb < W; kb += 32) {
      v16h a = load_frag_a(src, W, rb, kb, lane);
      v16h b = load_frag_b(mw.WoutT, W, 0, kb, lane);
      acc = __builtin_amdgcn_wmma_f32_16x16x32_f16(false, a, false, b, (short)0, acc, false, false);
    }
    const int n = lane & 15, hf = lane >> 4;
    if (n < OUT) {
      const float bv = mw.bout[n];
#pragma unroll
      for (int r = 0; r < 8; ++r) {
        int pi = base + rb + r + hf * 8;
        if (pi < total) {
          int b = (pi >= seg_len) ? 1 : 0;
          int j = pi - b * seg_len;
          out[((size_t)b * P + tmpl[j]) * 3 + n] = acc[r] + bv;
        }
      }
    }
  }
}

extern "C" void kernel_launch(void* const* d_in, const int* in_sizes, int n_in,
                              void* d_out, int out_size, void* d_ws, size_t ws_size,
                              hipStream_t stream) {
  (void)in_sizes; (void)n_in; (void)out_size; (void)ws_size;
  const int N = 30000, P = 150000;

  const float* verts   = (const float*)d_in[0];
  const float* coords  = (const float*)d_in[1];
  const float* bcoords = (const float*)d_in[2];
  const float* maps[3] = {(const float*)d_in[3], (const float*)d_in[4], (const float*)d_in[5]};
  const float* f1W = (const float*)d_in[78];
  const float* f1b = (const float*)d_in[79];
  const int* trilist  = (const int*)d_in[80];
  const int* tmpls[3] = {(const int*)d_in[81], (const int*)d_in[82], (const int*)d_in[83]};

  // Workspace layout: c40 | vf | lat | f16 weight arena
  char* ws = (char*)d_ws;
  float* c40 = (float*)ws; ws += (size_t)N * 40 * sizeof(float);
  float* vf  = (float*)ws; ws += (size_t)2 * N * 40 * sizeof(float);
  float* lat = (float*)ws; ws += (size_t)2 * N * 8 * sizeof(float);
  h16* arena = (h16*)ws;

  ConvJobs jobs{};
  int nj = 0;
  size_t off = 0;
  auto add = [&](int idx, int K, int Nsrc, int Ndst) -> const h16* {
    int Kp = PAD32(K);
    h16* dst = arena + off;
    jobs.j[nj++] = ConvJob{(const float*)d_in[idx], dst, K, Kp, Nsrc, Ndst};
    off += (size_t)Ndst * Kp;
    return dst;
  };

  // params flatten order (sorted keys): dec[0..2]{Wout,Ws[],bout,bs[]}, enc[0..2]{...}, f1W, f1b
  MlpW dec0{}, dec1{}, dec2{}, enc0{}, enc1{}, enc2{};
  {
    int dims[8] = {171, 256, 256, 256, 256, 427, 256, 256};
    for (int l = 0; l < 8; ++l) {
      dec0.Wt[l] = add(7 + l, dims[l], 256, 256);
      dec0.K[l] = PAD32(dims[l]);
      dec0.bias[l] = (const float*)d_in[16 + l];
    }
    dec0.WoutT = add(6, 256, 3, 16);
    dec0.bout = (const float*)d_in[15];
  }
  auto mk_dec64 = [&](int basei, MlpW& m) {
    int dims[6] = {171, 64, 64, 64, 235, 64};
    for (int l = 0; l < 6; ++l) {
      m.Wt[l] = add(basei + 1 + l, dims[l], 64, 64);
      m.K[l] = PAD32(dims[l]);
      m.bias[l] = (const float*)d_in[basei + 8 + l];
    }
    m.WoutT = add(basei, 64, 3, 16);
    m.bout = (const float*)d_in[basei + 7];
  };
  mk_dec64(24, dec1);
  mk_dec64(38, dec2);
  {
    int dims[4] = {80, 256, 256, 336};
    for (int l = 0; l < 4; ++l) {
      enc0.Wt[l] = add(53 + l, dims[l], 256, 256);
      enc0.K[l] = PAD32(dims[l]);
      enc0.bias[l] = (const float*)d_in[58 + l];
    }
    enc0.WoutT = add(52, 256, 8, 16);
    enc0.bout = (const float*)d_in[57];
  }
  auto mk_enc64 = [&](int basei, MlpW& m) {
    int dims[3] = {80, 64, 144};
    for (int l = 0; l < 3; ++l) {
      m.Wt[l] = add(basei + 1 + l, dims[l], 64, 64);
      m.K[l] = PAD32(dims[l]);
      m.bias[l] = (const float*)d_in[basei + 5 + l];
    }
    m.WoutT = add(basei, 64, 8, 16);
    m.bout = (const float*)d_in[basei + 4];
  };
  mk_enc64(62, enc1);
  mk_enc64(70, enc2);

  convert_weights<<<nj, 256, 0, stream>>>(jobs);
  prep_kernel<<<(N + 255) / 256, 256, 0, stream>>>(coords, verts, f1W, f1b, c40, vf, N);

  // encoders: segments [0,24000), [24000,27000), [27000,30000)
  enc_kernel<256, 4, 2><<<(2 * 24000 + 31) / 32, 128, 0, stream>>>(c40, vf, enc0, lat, 0, 24000, N);
  enc_kernel<64, 3, 1><<<(2 * 3000 + 31) / 32, 128, 0, stream>>>(c40, vf, enc1, lat, 24000, 3000, N);
  enc_kernel<64, 3, 1><<<(2 * 3000 + 31) / 32, 128, 0, stream>>>(c40, vf, enc2, lat, 27000, 3000, N);

  // decoders: segments [0,120000), [120000,135000), [135000,150000)
  float* out = (float*)d_out;
  dec_kernel<256, 8, 4><<<(2 * 120000 + 31) / 32, 128, 0, stream>>>(
      lat, c40, bcoords, trilist, maps[0], tmpls[0], dec0, out, 0, 120000, N, P);
  dec_kernel<64, 6, 3><<<(2 * 15000 + 31) / 32, 128, 0, stream>>>(
      lat, c40, bcoords, trilist, maps[1], tmpls[1], dec1, out, 120000, 15000, N, P);
  dec_kernel<64, 6, 3><<<(2 * 15000 + 31) / 32, 128, 0, stream>>>(
      lat, c40, bcoords, trilist, maps[2], tmpls[2], dec2, out, 135000, 15000, N, P);
}
